// MetaLSTMCell_86423331930501
// MI455X (gfx1250) — matrix operand order repaired and checked by reading, not compiled, not run on hardware
//
#include <hip/hip_runtime.h>
#include <hip/hip_bf16.h>
#include <math.h>

#define B_   2048
#define IN_  1024
#define H_   1024
#define Z_   256
#define G_   4
#define NZ_  1024   // 4*Z

typedef __attribute__((ext_vector_type(8)))  __bf16 v8bf;
typedef __attribute__((ext_vector_type(16))) __bf16 v16bf;
typedef __attribute__((ext_vector_type(8)))  float  v8f;

typedef __attribute__((ext_vector_type(4))) unsigned int u32x4;
typedef __attribute__((ext_vector_type(8))) int          i32x8;
typedef __attribute__((ext_vector_type(4))) int          i32x4;

#if defined(__has_include)
#  if __has_include(<hip/amd_detail/amd_gfx1250_TDM.h>)
#    define TDM_6ARG 1
#  else
#    define TDM_6ARG 0
#  endif
#else
#  define TDM_6ARG 0
#endif

// ---------------------------------------------------------------------------
// WMMA helpers (wave32, 16x16x32 bf16 -> f32 accumulate)
// ---------------------------------------------------------------------------
__device__ __forceinline__ v8f wmma_bf16(v16bf a, v16bf b, v8f c) {
  return __builtin_amdgcn_wmma_f32_16x16x32_bf16(
      /*neg_a=*/false, a, /*neg_b=*/false, b,
      /*c_mod=*/(short)0, c, /*reuse_a=*/false, /*reuse_b=*/false);
}

// A-matrix 16x32 bf16 fragment from row-major A (lda in elements).
// lanes 0-15: rows M=0..15 with K=0..15; lanes 16-31: K=16..31.
__device__ __forceinline__ v16bf load_a_frag(const __bf16* A, int lda) {
  const int lane = threadIdx.x & 31;
  const int m    = lane & 15;
  const int koff = (lane & 16) ? 16 : 0;
  const __bf16* p = A + (size_t)m * lda + koff;
  v8bf lo = *(const v8bf*)(p);
  v8bf hi = *(const v8bf*)(p + 8);
  v16bf r;
#pragma unroll
  for (int e = 0; e < 8; ++e) { r[e] = lo[e]; r[8 + e] = hi[e]; }
  return r;
}

// B-matrix 32x16 fragment where B = W^T, W row-major (N x K), from global.
__device__ __forceinline__ v16bf load_b_frag(const __bf16* W, int ldk) {
  const int lane = threadIdx.x & 31;
  const int col  = lane & 15;
  const int koff = (lane & 16) ? 16 : 0;
  return *(const v16bf*)(W + (size_t)col * ldk + koff);
}

// Same fragment but from an LDS-staged 64x32 row-major tile (T = tile + 16*j*32).
__device__ __forceinline__ v16bf load_b_frag_lds(const __bf16* T) {
  const int lane = threadIdx.x & 31;
  const int col  = lane & 15;
  const int koff = (lane & 16) ? 16 : 0;
  return *(const v16bf*)(T + col * 32 + koff);
}

// ---------------------------------------------------------------------------
// Tensor Data Mover: async-load a 64(row) x 32(col) bf16 tile (row stride ldk
// elements) from global into LDS at byte offset lds_off. Tracked by TENSORcnt.
// ---------------------------------------------------------------------------
__device__ __forceinline__ unsigned lds_addr_of(const void* p) {
  return (unsigned)(uintptr_t)p;   // low 32 bits of shared aperture = LDS addr
}

__device__ __forceinline__ void tdm_load_64x32_bf16(unsigned lds_off,
                                                    const __bf16* gptr,
                                                    int ldk_elems) {
  const unsigned long long ga = (unsigned long long)(uintptr_t)gptr;
  u32x4 g0;
  g0[0] = 1u;                                       // count=1, user descriptor
  g0[1] = lds_off;                                  // lds_addr (bytes)
  g0[2] = (unsigned)(ga & 0xffffffffull);           // global_addr[31:0]
  g0[3] = (unsigned)((ga >> 32) & 0x1ffffffull)     // global_addr[56:32]
          | (2u << 30);                             // type = 2 ("image")
  i32x8 g1;
  g1[0] = (int)(1u << 16);    // workgroup_mask=0, data_size=1 (2-byte elems)
  g1[1] = (int)(32u << 16);   // tensor_dim0 = 32 (row length, elems)
  g1[2] = (int)(64u << 16);   // tensor_dim1 = 64 (rows)
  g1[3] = (int)(32u << 16);   // tile_dim0   = 32
  g1[4] = 64;                 // tile_dim1   = 64, tile_dim2 = 0
  g1[5] = ldk_elems;          // tensor_dim0_stride[31:0]
  g1[6] = 0;                  // stride high bits / dim1_stride = 0
  g1[7] = 0;
  i32x4 z4 = {0, 0, 0, 0};
#if TDM_6ARG
  i32x8 z8 = {0, 0, 0, 0, 0, 0, 0, 0};
  __builtin_amdgcn_tensor_load_to_lds(g0, g1, z4, z4, z8, 0);
#else
  __builtin_amdgcn_tensor_load_to_lds(g0, g1, z4, z4, 0);
#endif
}

// ---------------------------------------------------------------------------
// Block-cooperative GEMM accumulation: 4 waves share TDM-staged B tiles in
// LDS (double buffered); each wave accumulates its own 16x64 C tile.
//   acc  : 4 x v8f accumulators (16x64 per wave)
//   A    : this wave's A rows (16 x ktot), lda elements
//   Wrow : W + n0*ldk (64 output rows of row-major N x K weight)
//   sbuf : LDS double buffer [2][64*32] bf16
// ---------------------------------------------------------------------------
__device__ __forceinline__ void mm_accum(v8f acc[4], const __bf16* A, int lda,
                                         const __bf16* Wrow, int ldk, int ktot,
                                         __bf16 (*sbuf)[64 * 32], bool issuer) {
  const int nch = ktot >> 5;
  if (issuer) tdm_load_64x32_bf16(lds_addr_of(sbuf[0]), Wrow, ldk);
  for (int i = 0; i < nch; ++i) {
    if (issuer) {
      if (i + 1 < nch) {
        tdm_load_64x32_bf16(lds_addr_of(sbuf[(i + 1) & 1]),
                            Wrow + (size_t)(i + 1) * 32, ldk);
        __builtin_amdgcn_s_wait_tensorcnt(1);   // chunk i complete (in-order)
      } else {
        __builtin_amdgcn_s_wait_tensorcnt(0);
      }
    }
    __syncthreads();                            // tile i visible to all waves
    __builtin_prefetch(A + (size_t)(i + 1) * 32, 0, 1);
    v16bf a = load_a_frag(A + (size_t)i * 32, lda);
    const __bf16* t = sbuf[i & 1];
#pragma unroll
    for (int j = 0; j < 4; ++j)
      acc[j] = wmma_bf16(a, load_b_frag_lds(t + 16 * j * 32), acc[j]);
    __syncthreads();                            // done reading buf[i&1]
  }
}

// ---------------------------------------------------------------------------
// Stage 1: f32 -> bf16 conversion (grid-stride)
// ---------------------------------------------------------------------------
__global__ void cvt_f32_bf16_kernel(const float* __restrict__ src,
                                    __bf16* __restrict__ dst, int n) {
  int i = blockIdx.x * blockDim.x + threadIdx.x;
  const int stride = gridDim.x * blockDim.x;
  for (; i < n; i += stride) dst[i] = (__bf16)src[i];
}

// ---------------------------------------------------------------------------
// Stage 2: z = meta(B x Z) @ W^T(Z x NZ) + bias  -> bf16 zout(B x NZ)
// ---------------------------------------------------------------------------
__global__ __launch_bounds__(128) void zgemm_kernel(
    const __bf16* __restrict__ meta, const __bf16* __restrict__ W,
    const float* __restrict__ bias, __bf16* __restrict__ zout) {
  const int wid   = blockIdx.x * 4 + (threadIdx.x >> 5);
  const int mtile = wid >> 4;
  const int ntile = wid & 15;
  const int m0 = mtile * 16, n0 = ntile * 64;
  const int lane = threadIdx.x & 31;

  v8f acc[4] = {};
  for (int k = 0; k < Z_; k += 32) {
    v16bf a = load_a_frag(meta + (size_t)m0 * Z_ + k, Z_);
#pragma unroll
    for (int j = 0; j < 4; ++j)
      acc[j] = wmma_bf16(a, load_b_frag(W + (size_t)(n0 + 16 * j) * Z_ + k, Z_),
                         acc[j]);
  }
  const int nn   = lane & 15;
  const int mrow = (lane & 16) ? 8 : 0;
#pragma unroll
  for (int j = 0; j < 4; ++j) {
    const int ncol = n0 + 16 * j + nn;
    const float bv = bias ? bias[ncol] : 0.0f;
#pragma unroll
    for (int r = 0; r < 8; ++r)
      zout[(size_t)(m0 + mrow + r) * NZ_ + ncol] = (__bf16)(acc[j][r] + bv);
  }
}

// ---------------------------------------------------------------------------
// Stage 3: fused  y[b,g,h] = d_h*wh + d_x*wx + d_b + db_b
// Block = 4 waves = 64x64 tile of (b,h) for one gate g; five GEMM
// accumulations share TDM-staged weight tiles in LDS.
// ---------------------------------------------------------------------------
__global__ __launch_bounds__(128) void fused_gemm_kernel(
    const __bf16* __restrict__ hbf,  const __bf16* __restrict__ xbf,
    const __bf16* __restrict__ zh,   const __bf16* __restrict__ zx,
    const __bf16* __restrict__ zb,
    const __bf16* __restrict__ whw,  const __bf16* __restrict__ wxw,
    const __bf16* __restrict__ dhw,  const __bf16* __restrict__ dxw,
    const __bf16* __restrict__ dbw,
    const float*  __restrict__ db_b, float* __restrict__ Y) {
  __shared__ __bf16 sbuf[2][64 * 32];   // double-buffered 64x32 B tile (8 KB)

  const int bid   = blockIdx.x;
  const int g     = bid >> 9;           // 32*16 = 512 blocks per gate
  const int rem   = bid & 511;
  const int mtile = rem >> 4;           // 32 m-tiles of 64 rows
  const int ntile = rem & 15;           // 16 n-tiles of 64 cols
  const int warp  = threadIdx.x >> 5;
  const int m0    = mtile * 64 + warp * 16;   // this wave's 16 rows
  const int n0    = ntile * 64;
  const int lane  = threadIdx.x & 31;
  const bool issuer = (warp == 0);

  v8f res[4], t1[4], t2[4];
#pragma unroll
  for (int j = 0; j < 4; ++j) { t1[j] = (v8f){}; t2[j] = (v8f){}; }

  // wh = h @ w_h[g]^T (K=H)  and  d_h = zh_g @ dh_w[g]^T (K=Z)
  mm_accum(t1, hbf + (size_t)m0 * H_, H_,
           whw + ((size_t)g * H_ + n0) * H_, H_, H_, sbuf, issuer);
  mm_accum(t2, zh + (size_t)m0 * NZ_ + g * Z_, NZ_,
           dhw + ((size_t)g * H_ + n0) * Z_, Z_, Z_, sbuf, issuer);
#pragma unroll
  for (int j = 0; j < 4; ++j) { res[j] = t1[j] * t2[j]; t1[j] = (v8f){}; t2[j] = (v8f){}; }

  // wx = x @ w_x[g]^T (K=IN)  and  d_x = zx_g @ dx_w[g]^T (K=Z)
  mm_accum(t1, xbf + (size_t)m0 * IN_, IN_,
           wxw + ((size_t)g * H_ + n0) * IN_, IN_, IN_, sbuf, issuer);
  mm_accum(t2, zx + (size_t)m0 * NZ_ + g * Z_, NZ_,
           dxw + ((size_t)g * H_ + n0) * Z_, Z_, Z_, sbuf, issuer);
#pragma unroll
  for (int j = 0; j < 4; ++j) { res[j] += t1[j] * t2[j]; t1[j] = (v8f){}; }

  // d_b = zb_g @ db_w[g]^T (K=Z)
  mm_accum(t1, zb + (size_t)m0 * NZ_ + g * Z_, NZ_,
           dbw + ((size_t)g * H_ + n0) * Z_, Z_, Z_, sbuf, issuer);
#pragma unroll
  for (int j = 0; j < 4; ++j) res[j] += t1[j];

  // bias + store y (f32, layout B x G x H)
  const int nn   = lane & 15;
  const int mrow = (lane & 16) ? 8 : 0;
#pragma unroll
  for (int j = 0; j < 4; ++j) {
    const int hcol = n0 + 16 * j + nn;
    const float bv = db_b[g * H_ + hcol];
#pragma unroll
    for (int r = 0; r < 8; ++r)
      Y[(size_t)(m0 + mrow + r) * (G_ * H_) + (size_t)g * H_ + hcol] =
          res[j][r] + bv;
  }
}

// ---------------------------------------------------------------------------
// Stage 4: per-row LayerNorm (over H, per gate) + LSTM gate math.
// ---------------------------------------------------------------------------
__device__ __forceinline__ float sigm(float x) { return 1.0f / (1.0f + expf(-x)); }

__global__ __launch_bounds__(256) void ln_gates_kernel(
    const float* __restrict__ Y, const float* __restrict__ c,
    const float* __restrict__ ln_w, const float* __restrict__ ln_b,
    float* __restrict__ out) {
  const int b   = blockIdx.x;
  const int tid = threadIdx.x;
  __shared__ float red[256];

  float yn[G_][4];
#pragma unroll
  for (int g = 0; g < G_; ++g) {
    const float* row = Y + ((size_t)b * G_ + g) * H_;
    float v[4];
    float s = 0.0f, s2 = 0.0f;
#pragma unroll
    for (int q = 0; q < 4; ++q) {
      v[q] = row[tid + 256 * q];
      s  += v[q];
      s2 += v[q] * v[q];
    }
    red[tid] = s;
    __syncthreads();
    for (int off = 128; off > 0; off >>= 1) {
      if (tid < off) red[tid] += red[tid + off];
      __syncthreads();
    }
    const float mu = red[0] * (1.0f / H_);
    __syncthreads();
    red[tid] = s2;
    __syncthreads();
    for (int off = 128; off > 0; off >>= 1) {
      if (tid < off) red[tid] += red[tid + off];
      __syncthreads();
    }
    const float var = red[0] * (1.0f / H_) - mu * mu;
    __syncthreads();
    const float inv = rsqrtf(var + 1e-5f);
#pragma unroll
    for (int q = 0; q < 4; ++q) {
      const int hh = tid + 256 * q;
      yn[g][q] = (v[q] - mu) * inv * ln_w[g * H_ + hh] + ln_b[g * H_ + hh];
    }
  }

#pragma unroll
  for (int q = 0; q < 4; ++q) {
    const int hh = tid + 256 * q;
    const float iv = yn[0][q], fv = yn[1][q], gv = yn[2][q], ov = yn[3][q];
    const float cv = c[(size_t)b * H_ + hh];
    const float cn = sigm(fv) * cv + sigm(iv) * tanhf(gv);
    const float hn = sigm(ov) * tanhf(cn);
    out[(size_t)b * H_ + hh]                   = hn;
    out[(size_t)B_ * H_ + (size_t)b * H_ + hh] = cn;
  }
}

// ---------------------------------------------------------------------------
// Host launcher
// ---------------------------------------------------------------------------
extern "C" void kernel_launch(void* const* d_in, const int* in_sizes, int n_in,
                              void* d_out, int out_size, void* d_ws, size_t ws_size,
                              hipStream_t stream) {
  const float* src_x = (const float*)d_in[0];
  const float* h_in  = (const float*)d_in[1];
  const float* c_in  = (const float*)d_in[2];
  const float* meta  = (const float*)d_in[3];
  const float* zh_w  = (const float*)d_in[4];
  const float* zh_b  = (const float*)d_in[5];
  const float* zx_w  = (const float*)d_in[6];
  const float* zx_b  = (const float*)d_in[7];
  const float* zb_w  = (const float*)d_in[8];
  const float* dh_w  = (const float*)d_in[9];
  const float* dx_w  = (const float*)d_in[10];
  const float* db_w  = (const float*)d_in[11];
  const float* db_b  = (const float*)d_in[12];
  const float* w_h   = (const float*)d_in[13];
  const float* w_x   = (const float*)d_in[14];
  const float* ln_w  = (const float*)d_in[15];
  const float* ln_b  = (const float*)d_in[16];
  float* out = (float*)d_out;

  char* base = (char*)d_ws;
  size_t off = 0;
  auto alloc = [&](size_t bytes) -> char* {
    char* p = base + off;
    off += (bytes + 255) & ~(size_t)255;
    return p;
  };
  float*  Y      = (float*) alloc((size_t)B_ * G_ * H_ * sizeof(float));
  __bf16* hbf    = (__bf16*)alloc((size_t)B_ * H_  * 2);
  __bf16* xbf    = (__bf16*)alloc((size_t)B_ * IN_ * 2);
  __bf16* metabf = (__bf16*)alloc((size_t)B_ * Z_  * 2);
  __bf16* zhbf   = (__bf16*)alloc((size_t)B_ * NZ_ * 2);
  __bf16* zxbf   = (__bf16*)alloc((size_t)B_ * NZ_ * 2);
  __bf16* zbbf   = (__bf16*)alloc((size_t)B_ * NZ_ * 2);
  __bf16* whwbf  = (__bf16*)alloc((size_t)G_ * H_ * H_  * 2);
  __bf16* wxwbf  = (__bf16*)alloc((size_t)G_ * H_ * IN_ * 2);
  __bf16* dhwbf  = (__bf16*)alloc((size_t)G_ * H_ * Z_  * 2);
  __bf16* dxwbf  = (__bf16*)alloc((size_t)G_ * H_ * Z_  * 2);
  __bf16* dbwbf  = (__bf16*)alloc((size_t)G_ * H_ * Z_  * 2);
  __bf16* zhwbf  = (__bf16*)alloc((size_t)NZ_ * Z_ * 2);
  __bf16* zxwbf  = (__bf16*)alloc((size_t)NZ_ * Z_ * 2);
  __bf16* zbwbf  = (__bf16*)alloc((size_t)NZ_ * Z_ * 2);
  (void)ws_size; (void)in_sizes; (void)n_in; (void)out_size;

  auto cvt = [&](const float* s, __bf16* d, int n) {
    cvt_f32_bf16_kernel<<<1024, 256, 0, stream>>>(s, d, n);
  };
  cvt(h_in,  hbf,    B_ * H_);
  cvt(src_x, xbf,    B_ * IN_);
  cvt(meta,  metabf, B_ * Z_);
  cvt(w_h,   whwbf,  G_ * H_ * H_);
  cvt(w_x,   wxwbf,  G_ * H_ * IN_);
  cvt(dh_w,  dhwbf,  G_ * H_ * Z_);
  cvt(dx_w,  dxwbf,  G_ * H_ * Z_);
  cvt(db_w,  dbwbf,  G_ * H_ * Z_);
  cvt(zh_w,  zhwbf,  NZ_ * Z_);
  cvt(zx_w,  zxwbf,  NZ_ * Z_);
  cvt(zb_w,  zbwbf,  NZ_ * Z_);

  zgemm_kernel<<<512, 128, 0, stream>>>(metabf, zhwbf, zh_b,    zhbf);
  zgemm_kernel<<<512, 128, 0, stream>>>(metabf, zxwbf, zx_b,    zxbf);
  zgemm_kernel<<<512, 128, 0, stream>>>(metabf, zbwbf, nullptr, zbbf);

  fused_gemm_kernel<<<2048, 128, 0, stream>>>(hbf, xbf, zhbf, zxbf, zbbf,
                                              whwbf, wxwbf, dhwbf, dxwbf, dbwbf,
                                              db_b, Y);

  ln_gates_kernel<<<B_, 256, 0, stream>>>(Y, c_in, ln_w, ln_b, out);
}
